// AttnDecoder1_33088428048482
// MI455X (gfx1250) — matrix operand order repaired and checked by reading, not compiled, not run on hardware
//
#include <hip/hip_runtime.h>
#include <hip/hip_bf16.h>
#include <math.h>

// ---------------------------------------------------------------------------
// Problem sizes (fixed by the reference)
// ---------------------------------------------------------------------------
#define BB 64      // batch
#define TT 128     // decode steps
#define HH 1024    // hidden
#define LL 512     // max length
#define K2 2048    // 2*H

typedef __bf16 bf16_t;
typedef __attribute__((ext_vector_type(16))) __bf16 v16bf;
typedef __attribute__((ext_vector_type(8)))  float  v8f;

// ---------------------------------------------------------------------------
// Workspace layout (bytes).  Total ~13.3 MB.
// ---------------------------------------------------------------------------
#define OFF_WATTN   0                                   // [512 ,2048] bf16
#define OFF_WCOMB   (OFF_WATTN + 512*2048*2)            // [1024,2048] bf16
#define OFF_WIHHH   (OFF_WCOMB + 1024*2048*2)           // [1024,2048] bf16 ([W_ih|W_hh])
#define OFF_WOUT    (OFF_WIHHH + 1024*2048*2)           // [1024,1024] bf16
#define OFF_BSUM    (OFF_WOUT  + 1024*1024*2)           // [1024] f32 (b_ih+b_hh)
#define OFF_LOGIT   (OFF_BSUM  + 1024*4)                // [64,512]  f32
#define OFF_APPL    (OFF_LOGIT + BB*LL*4)               // [64,1024] f32 (attention applied)
#define OFF_COMB    (OFF_APPL  + BB*HH*4)               // [64,1024] f32
#define OFF_HBUF    (OFF_COMB  + BB*HH*4)               // [64,1024] f32 (carried h)
#define OFF_HNEW    (OFF_HBUF  + BB*HH*4)               // [64,1024] f32
#define OFF_STOP    (OFF_HNEW  + BB*HH*4)               // [64] int

// ---------------------------------------------------------------------------
// Weight conversion / setup kernels (run every launch; deterministic)
// ---------------------------------------------------------------------------
__global__ void cvt_bf16_kernel(const float* __restrict__ src,
                                bf16_t* __restrict__ dst, int n) {
    for (int i = blockIdx.x * blockDim.x + threadIdx.x; i < n;
         i += gridDim.x * blockDim.x)
        dst[i] = (bf16_t)src[i];
}

// dst[n][k] = k<H ? W_ih[n][k] : W_hh[n][k-H]   (shape [1024, 2048])
__global__ void build_ihhh_kernel(const float* __restrict__ Wih,
                                  const float* __restrict__ Whh,
                                  bf16_t* __restrict__ dst) {
    const int n = HH * K2;
    for (int i = blockIdx.x * blockDim.x + threadIdx.x; i < n;
         i += gridDim.x * blockDim.x) {
        int row = i >> 11;          // /2048
        int k   = i & (K2 - 1);
        float v = (k < HH) ? Wih[row * HH + k] : Whh[row * HH + (k - HH)];
        dst[i] = (bf16_t)v;
    }
}

__global__ void bias_sum_kernel(const float* __restrict__ bih,
                                const float* __restrict__ bhh,
                                float* __restrict__ bsum) {
    int i = blockIdx.x * blockDim.x + threadIdx.x;
    if (i < HH) bsum[i] = bih[i] + bhh[i];
}

// hbuf = decode_hidden ; stop = 0
__global__ void init_kernel(const float* __restrict__ h0,
                            float* __restrict__ hbuf, int* __restrict__ stop) {
    int i = blockIdx.x * blockDim.x + threadIdx.x;
    if (i < BB * HH) hbuf[i] = h0[i];
    if (i < BB) stop[i] = 0;
}

// ---------------------------------------------------------------------------
// Softmax over logits[b][:] -> applied = aw @ enc[b]; also updates stop flag
// from mean(x_t) == 0.   grid = 64 blocks (one sample), 256 threads
// ---------------------------------------------------------------------------
__global__ void softmax_apply_stop_kernel(const float* __restrict__ logits,
                                          const float* __restrict__ enc,
                                          const float* __restrict__ dec_in, int t,
                                          float* __restrict__ applied,
                                          int* __restrict__ stop) {
    const int b = blockIdx.x, tid = threadIdx.x;
    __shared__ float aw[LL];
    __shared__ float red[256];

    // ---- stop flag: stop |= (mean(x_t) == 0)
    const float* x = dec_in + ((size_t)b * TT + t) * HH;
    float s = 0.f;
    for (int c = tid; c < HH; c += 256) s += x[c];
    red[tid] = s;
    __syncthreads();
    for (int off = 128; off > 0; off >>= 1) {
        if (tid < off) red[tid] += red[tid + off];
        __syncthreads();
    }
    if (tid == 0 && red[0] * (1.0f / (float)HH) == 0.0f) stop[b] = 1;
    __syncthreads();

    // ---- softmax over 512 logits
    float v0 = logits[b * LL + tid];
    float v1 = logits[b * LL + tid + 256];
    red[tid] = fmaxf(v0, v1);
    __syncthreads();
    for (int off = 128; off > 0; off >>= 1) {
        if (tid < off) red[tid] = fmaxf(red[tid], red[tid + off]);
        __syncthreads();
    }
    const float mx = red[0];
    __syncthreads();
    float e0 = __expf(v0 - mx), e1 = __expf(v1 - mx);
    aw[tid] = e0; aw[tid + 256] = e1;
    red[tid] = e0 + e1;
    __syncthreads();
    for (int off = 128; off > 0; off >>= 1) {
        if (tid < off) red[tid] += red[tid + off];
        __syncthreads();
    }
    const float inv = 1.0f / red[0];
    __syncthreads();
    aw[tid] *= inv; aw[tid + 256] *= inv;
    __syncthreads();

    // ---- applied = aw @ enc[b]   (per-sample GEMV; weights differ per b)
    const float* eb = enc + (size_t)b * LL * HH;
    for (int c = tid; c < HH; c += 256) {
        float acc = 0.f;
        #pragma unroll 4
        for (int l = 0; l < LL; ++l)
            acc = fmaf(aw[l], eb[(size_t)l * HH + c], acc);
        applied[b * HH + c] = acc;
    }
}

// ---------------------------------------------------------------------------
// K-chunk accumulate: 16x32-per-step bf16 WMMA over one A source.
//   arow : per-lane A row pointer (f32), already offset by half*16
//   b0..b3: per-lane weight row pointers (bf16, [N,K] row-major),
//           already offset by half*16 and the chunk's K base
// ---------------------------------------------------------------------------
__device__ __forceinline__ void wmma_kchunk(const float* __restrict__ arow,
                                            const bf16_t* __restrict__ b0p,
                                            const bf16_t* __restrict__ b1p,
                                            const bf16_t* __restrict__ b2p,
                                            const bf16_t* __restrict__ b3p,
                                            int kcount,
                                            v8f& acc0, v8f& acc1,
                                            v8f& acc2, v8f& acc3) {
    for (int kk = 0; kk < kcount; kk += 32) {
        // A fragment: lane holds 16 contiguous K values {h*16 .. h*16+15}
        float4 f0 = *(const float4*)(arow + kk);
        float4 f1 = *(const float4*)(arow + kk + 4);
        float4 f2 = *(const float4*)(arow + kk + 8);
        float4 f3 = *(const float4*)(arow + kk + 12);
        v16bf a;
        a[0]  = (bf16_t)f0.x; a[1]  = (bf16_t)f0.y; a[2]  = (bf16_t)f0.z; a[3]  = (bf16_t)f0.w;
        a[4]  = (bf16_t)f1.x; a[5]  = (bf16_t)f1.y; a[6]  = (bf16_t)f1.z; a[7]  = (bf16_t)f1.w;
        a[8]  = (bf16_t)f2.x; a[9]  = (bf16_t)f2.y; a[10] = (bf16_t)f2.z; a[11] = (bf16_t)f2.w;
        a[12] = (bf16_t)f3.x; a[13] = (bf16_t)f3.y; a[14] = (bf16_t)f3.z; a[15] = (bf16_t)f3.w;

        // B fragments: lane holds 16 contiguous K values at fixed N
        v16bf b0 = *(const v16bf*)(b0p + kk);
        v16bf b1 = *(const v16bf*)(b1p + kk);
        v16bf b2 = *(const v16bf*)(b2p + kk);
        v16bf b3 = *(const v16bf*)(b3p + kk);

        acc0 = __builtin_amdgcn_wmma_f32_16x16x32_bf16(false, a, false, b0, (short)0, acc0, false, false);
        acc1 = __builtin_amdgcn_wmma_f32_16x16x32_bf16(false, a, false, b1, (short)0, acc1, false, false);
        acc2 = __builtin_amdgcn_wmma_f32_16x16x32_bf16(false, a, false, b2, (short)0, acc2, false, false);
        acc3 = __builtin_amdgcn_wmma_f32_16x16x32_bf16(false, a, false, b3, (short)0, acc3, false, false);
    }
}

// ---------------------------------------------------------------------------
// WMMA GEMM with virtual K-concat A:
//   C[64, N] = act( [A0 | A1] @ Wb^T + bias ),  K = K0 + K1
//   A0/A1: f32 row-major (lda0/lda1)  — concat never materialized
//   Wb   : bf16 [N, K] row-major (rows are WMMA B-fragment layout directly)
//   block = 256 threads = 8 wave32s; wave tile 16M x 64N; WG tile 64M x 128N
//   gridDim.x = N / 128
// ACT: 0 = store, 1 = relu, 2 = tanh, 3 = stop-mask store + h commit
// ---------------------------------------------------------------------------
template <int ACT>
__global__ void gemm_wmma_kernel(const float* __restrict__ A0, int lda0, int K0,
                                 const float* __restrict__ A1, int lda1, int K1,
                                 const bf16_t* __restrict__ Wb,
                                 const float* __restrict__ bias,
                                 float* __restrict__ C, int ldc,
                                 const int* __restrict__ stop,
                                 const float* __restrict__ hsrc,
                                 float* __restrict__ hdst) {
    const int lane = threadIdx.x & 31;
    const int wave = threadIdx.x >> 5;        // 0..7
    const int half = lane >> 4;               // 0/1
    const int r    = lane & 15;

    const int m0    = (wave & 3) * 16;                       // 0,16,32,48
    const int nbase = blockIdx.x * 128 + (wave >> 2) * 64;   // wave N offset
    const int K     = K0 + K1;

    v8f acc0 = {}, acc1 = {}, acc2 = {}, acc3 = {};

    const bf16_t* brow0 = Wb + (size_t)(nbase +  0 + r) * K + half * 16;
    const bf16_t* brow1 = Wb + (size_t)(nbase + 16 + r) * K + half * 16;
    const bf16_t* brow2 = Wb + (size_t)(nbase + 32 + r) * K + half * 16;
    const bf16_t* brow3 = Wb + (size_t)(nbase + 48 + r) * K + half * 16;

    // first K chunk from A0
    {
        const float* arow = A0 + (size_t)(m0 + r) * lda0 + half * 16;
        wmma_kchunk(arow, brow0, brow1, brow2, brow3, K0, acc0, acc1, acc2, acc3);
    }
    // second K chunk from A1 (weight rows advance by K0)
    if (K1 > 0) {
        const float* arow = A1 + (size_t)(m0 + r) * lda1 + half * 16;
        wmma_kchunk(arow, brow0 + K0, brow1 + K0, brow2 + K0, brow3 + K0, K1,
                    acc0, acc1, acc2, acc3);
    }

    // epilogue: acc[i] holds C[m0 + 8*half + i][nsub + r]
    v8f accs[4] = {acc0, acc1, acc2, acc3};
    #pragma unroll
    for (int s = 0; s < 4; ++s) {
        const int col  = nbase + s * 16 + r;
        const float bv = bias[col];
        #pragma unroll
        for (int i = 0; i < 8; ++i) {
            const int m = m0 + half * 8 + i;
            float v = accs[s][i] + bv;
            if (ACT == 1) v = v > 0.f ? v : 0.f;
            else if (ACT == 2) v = tanhf(v);
            else if (ACT == 3) {
                if (stop[m] != 0) v = 0.f;
                else hdst[(size_t)m * HH + col] = hsrc[(size_t)m * HH + col];
            }
            C[(size_t)m * ldc + col] = v;
        }
    }
}

// ---------------------------------------------------------------------------
// Launch
// ---------------------------------------------------------------------------
extern "C" void kernel_launch(void* const* d_in, const int* in_sizes, int n_in,
                              void* d_out, int out_size, void* d_ws, size_t ws_size,
                              hipStream_t stream) {
    const float* dec_in = (const float*)d_in[0];
    const float* dec_h  = (const float*)d_in[1];
    const float* enc    = (const float*)d_in[2];
    const float* W_attn = (const float*)d_in[3];
    const float* b_attn = (const float*)d_in[4];
    const float* W_comb = (const float*)d_in[5];
    const float* b_comb = (const float*)d_in[6];
    const float* W_ih   = (const float*)d_in[7];
    const float* b_ih   = (const float*)d_in[8];
    const float* W_hh   = (const float*)d_in[9];
    const float* b_hh   = (const float*)d_in[10];
    const float* W_out  = (const float*)d_in[11];
    const float* b_out  = (const float*)d_in[12];
    float* out = (float*)d_out;

    char* ws = (char*)d_ws;
    bf16_t* wattn   = (bf16_t*)(ws + OFF_WATTN);
    bf16_t* wcomb   = (bf16_t*)(ws + OFF_WCOMB);
    bf16_t* wihhh   = (bf16_t*)(ws + OFF_WIHHH);
    bf16_t* wout    = (bf16_t*)(ws + OFF_WOUT);
    float*  bsum    = (float*) (ws + OFF_BSUM);
    float*  logit   = (float*) (ws + OFF_LOGIT);
    float*  applied = (float*) (ws + OFF_APPL);
    float*  comb    = (float*) (ws + OFF_COMB);
    float*  hbuf    = (float*) (ws + OFF_HBUF);
    float*  hnew    = (float*) (ws + OFF_HNEW);
    int*    stopf   = (int*)   (ws + OFF_STOP);

    // weight conversion (bf16) + bias fuse + state init
    cvt_bf16_kernel<<<512, 256, 0, stream>>>(W_attn, wattn, LL * K2);
    cvt_bf16_kernel<<<512, 256, 0, stream>>>(W_comb, wcomb, HH * K2);
    cvt_bf16_kernel<<<512, 256, 0, stream>>>(W_out,  wout,  HH * HH);
    build_ihhh_kernel<<<512, 256, 0, stream>>>(W_ih, W_hh, wihhh);
    bias_sum_kernel<<<4, 256, 0, stream>>>(b_ih, b_hh, bsum);
    init_kernel<<<(BB * HH + 255) / 256, 256, 0, stream>>>(dec_h, hbuf, stopf);

    for (int t = 0; t < TT; ++t) {
        const float* xt = dec_in + (size_t)t * HH;   // row b at stride T*H

        // logits = [x_t | h] @ W_attn^T + b_attn        [64,512], K=1024+1024
        gemm_wmma_kernel<0><<<LL / 128, 256, 0, stream>>>(
            xt, TT * HH, HH, hbuf, HH, HH,
            wattn, b_attn, logit, LL, nullptr, nullptr, nullptr);

        // softmax + applied (per-sample GEMV) + stop update
        softmax_apply_stop_kernel<<<BB, 256, 0, stream>>>(
            logit, enc, dec_in, t, applied, stopf);

        // comb = relu([x_t | applied] @ W_comb^T + b_comb)   [64,1024]
        gemm_wmma_kernel<1><<<HH / 128, 256, 0, stream>>>(
            xt, TT * HH, HH, applied, HH, HH,
            wcomb, b_comb, comb, HH, nullptr, nullptr, nullptr);

        // h_new = tanh([comb | h] @ [W_ih|W_hh]^T + (b_ih+b_hh))   [64,1024]
        gemm_wmma_kernel<2><<<HH / 128, 256, 0, stream>>>(
            comb, HH, HH, hbuf, HH, HH,
            wihhh, bsum, hnew, HH, nullptr, nullptr, nullptr);

        // out_t = mask(h_new @ W_out^T + b_out); fused h-commit (h = stop? h : h_new)
        gemm_wmma_kernel<3><<<HH / 128, 256, 0, stream>>>(
            hnew, HH, HH, hnew, HH, 0,
            wout, b_out, out + (size_t)t * HH, TT * HH, stopf, hnew, hbuf);
    }
}
